// GlobalPointer_89421219103176
// MI455X (gfx1250) — matrix operand order repaired and checked by reading, not compiled, not run on hardware
//
#include <hip/hip_runtime.h>
#include <hip/hip_bf16.h>

// ---------------- problem constants ----------------
#define ENT     9
#define INNER   64
#define S_LEN   512
#define B_SZ    16
#define HID     768
#define OUTD    1152          // ENT * INNER * 2
#define NEGINF  1000000000000.0f

typedef __attribute__((ext_vector_type(16))) __bf16 v16bf;
typedef __attribute__((ext_vector_type(8)))  float  v8f;

union Frag { v16bf v; unsigned u[8]; };
union Acc  { v8f   v; float    f[8]; };

static __device__ __forceinline__ unsigned short f2bf(float f) {
  unsigned u = __float_as_uint(f);
  u += 0x7FFFu + ((u >> 16) & 1u);       // round-to-nearest-even
  return (unsigned short)(u >> 16);
}
static __device__ __forceinline__ float bf2f(unsigned short h) {
  return __uint_as_float(((unsigned)h) << 16);
}
static __device__ __forceinline__ v8f wmma_bf16(const v16bf& a, const v16bf& b, v8f c) {
  return __builtin_amdgcn_wmma_f32_16x16x32_bf16(false, a, false, b, (short)0, c, false, false);
}

// =====================================================================
// Kernel 1: proj = X @ W + b  (bf16x3 split WMMA), fused RoPE epilogue.
//   Block tile: 128(M) x 64(N), 8 waves of 32x32, K-chunks of 32.
//   Each block's 64 N-columns are exactly the q-half or k-half of one head.
//   Writes q/k as bf16 hi/lo pairs in (b, head, s, d) layout (packed u32).
// =====================================================================
__global__ __launch_bounds__(256) void proj_rope_kernel(
    const float* __restrict__ X,     // (8192, 768)
    const float* __restrict__ W,     // (768, 1152)
    const float* __restrict__ bias,  // (1152)
    unsigned* __restrict__ qhi, unsigned* __restrict__ qlo,
    unsigned* __restrict__ khi, unsigned* __restrict__ klo)
{
  __shared__ __align__(16) char smem[32768];
  unsigned short* sXhi = (unsigned short*)smem;        // [128][36] bf16
  unsigned short* sXlo = sXhi + 128 * 36;
  unsigned short* sWhi = sXlo + 128 * 36;              // [64][36]  W^T: [n][k]
  unsigned short* sWlo = sWhi + 64 * 36;

  const int t    = threadIdx.x;
  const int lane = t & 31;
  const int ln   = lane & 15;
  const int lh   = lane >> 4;
  const int wid  = t >> 5;
  const int wm   = wid >> 1;           // 0..3 : M offset wm*32
  const int wn   = wid & 1;            // 0..1 : N offset wn*32
  const int bm   = blockIdx.y;         // M tile: rows [bm*128, +128)
  const int nb   = blockIdx.x;         // N tile: cols [nb*64, +64)

  v8f zero = {0.f,0.f,0.f,0.f,0.f,0.f,0.f,0.f};
  Acc acc[2][2];
#pragma unroll
  for (int i = 0; i < 2; ++i)
#pragma unroll
    for (int j = 0; j < 2; ++j) acc[i][j].v = zero;

  for (int kc = 0; kc < HID; kc += 32) {
    __syncthreads();
    // ---- stage X tile (128 x 32) as hi/lo bf16 ----
#pragma unroll
    for (int j = 0; j < 4; ++j) {
      int q   = t + j * 256;           // 1024 float4 total
      int row = q >> 3;
      int c4  = (q & 7) * 4;
      float4 xv = *(const float4*)(X + (size_t)(bm * 128 + row) * HID + kc + c4);
      unsigned short hx = f2bf(xv.x), hy = f2bf(xv.y), hz = f2bf(xv.z), hw = f2bf(xv.w);
      unsigned short lx = f2bf(xv.x - bf2f(hx)), ly = f2bf(xv.y - bf2f(hy));
      unsigned short lz = f2bf(xv.z - bf2f(hz)), lw = f2bf(xv.w - bf2f(hw));
      unsigned* ph = (unsigned*)sXhi;
      unsigned* pl = (unsigned*)sXlo;
      int base = row * 18 + (c4 >> 1);
      ph[base]     = (unsigned)hx | ((unsigned)hy << 16);
      ph[base + 1] = (unsigned)hz | ((unsigned)hw << 16);
      pl[base]     = (unsigned)lx | ((unsigned)ly << 16);
      pl[base + 1] = (unsigned)lz | ((unsigned)lw << 16);
    }
    // ---- stage W tile (32 x 64), transposed to [n][k] ----
#pragma unroll
    for (int j = 0; j < 2; ++j) {
      int q  = t + j * 256;            // 512 float4 total
      int kr = q >> 4;                 // 0..31
      int c4 = (q & 15) * 4;           // 0..60
      float4 wv = *(const float4*)(W + (size_t)(kc + kr) * OUTD + nb * 64 + c4);
      float e[4] = {wv.x, wv.y, wv.z, wv.w};
#pragma unroll
      for (int s = 0; s < 4; ++s) {
        unsigned short h = f2bf(e[s]);
        unsigned short l = f2bf(e[s] - bf2f(h));
        sWhi[(c4 + s) * 36 + kr] = h;
        sWlo[(c4 + s) * 36 + kr] = l;
      }
    }
    __syncthreads();

    // ---- fragment gathers (ISA 7.12.2 layouts) ----
    Frag ahi[2], alo[2], bhi[2], blo[2];
#pragma unroll
    for (int mt = 0; mt < 2; ++mt) {
      int m = wm * 32 + mt * 16 + ln;
#pragma unroll
      for (int v = 0; v < 8; ++v) {
        int k = ((v & 4) << 2) + lh * 8 + ((v & 3) << 1);   // A: K=(v<4?0:16)+h*8+2*(v%4)
        ahi[mt].u[v] = ((const unsigned*)sXhi)[m * 18 + (k >> 1)];
        alo[mt].u[v] = ((const unsigned*)sXlo)[m * 18 + (k >> 1)];
      }
    }
#pragma unroll
    for (int nt = 0; nt < 2; ++nt) {
      int n = wn * 32 + nt * 16 + ln;
#pragma unroll
      for (int v = 0; v < 8; ++v) {
        int k = ((v & 4) << 2) + lh * 8 + ((v & 3) << 1);   // B: K=(v<4?0:16)+h*8+2*(v%4)
        bhi[nt].u[v] = ((const unsigned*)sWhi)[n * 18 + (k >> 1)];
        blo[nt].u[v] = ((const unsigned*)sWlo)[n * 18 + (k >> 1)];
      }
    }
    // ---- bf16x3: AhBh + AhBl + AlBh ----
#pragma unroll
    for (int mt = 0; mt < 2; ++mt)
#pragma unroll
      for (int nt = 0; nt < 2; ++nt) {
        acc[mt][nt].v = wmma_bf16(ahi[mt].v, bhi[nt].v, acc[mt][nt].v);
        acc[mt][nt].v = wmma_bf16(ahi[mt].v, blo[nt].v, acc[mt][nt].v);
        acc[mt][nt].v = wmma_bf16(alo[mt].v, bhi[nt].v, acc[mt][nt].v);
      }
  }

  // ---- spill accumulators to LDS as f32 tile [128][64] ----
  __syncthreads();
  float* sOut = (float*)smem;
#pragma unroll
  for (int mt = 0; mt < 2; ++mt)
#pragma unroll
    for (int nt = 0; nt < 2; ++nt)
#pragma unroll
      for (int r = 0; r < 8; ++r) {
        int m = wm * 32 + mt * 16 + r + 8 * lh;             // C/D: M=r+8*h, N=lane&15
        int n = wn * 32 + nt * 16 + ln;
        sOut[m * 64 + n] = acc[mt][nt].f[r];
      }
  __syncthreads();

  // ---- bias + interleaved RoPE + hi/lo split store ----
  const int head = nb >> 1;
  const bool is_q = (nb & 1) == 0;
  unsigned* dhi = is_q ? qhi : khi;
  unsigned* dlo = is_q ? qlo : klo;
#pragma unroll
  for (int j = 0; j < 16; ++j) {
    int p   = t + j * 256;             // 4096 (row, pair) items
    int row = p >> 5;                  // 0..127
    int i   = p & 31;                  // pair index -> d = 2i, 2i+1
    int gm  = bm * 128 + row;
    int b   = gm >> 9;
    int s   = gm & 511;
    float x0 = sOut[row * 64 + 2 * i]     + bias[nb * 64 + 2 * i];
    float x1 = sOut[row * 64 + 2 * i + 1] + bias[nb * 64 + 2 * i + 1];
    // freq = s * 10000^(-2i/64) = s * 2^(-i * log2(10000)/32)
    float fr = (float)s * exp2f(-(float)i * 0.41524101186f);
    float sn, cs;
    __sincosf(fr, &sn, &cs);
    float y0 = x0 * cs - x1 * sn;
    float y1 = x1 * cs + x0 * sn;
    unsigned short h0 = f2bf(y0), h1 = f2bf(y1);
    unsigned short l0 = f2bf(y0 - bf2f(h0)), l1 = f2bf(y1 - bf2f(h1));
    size_t idx = ((size_t)(b * ENT + head) * S_LEN + s) * 32 + i;  // u32 units (d pairs)
    dhi[idx] = (unsigned)h0 | ((unsigned)h1 << 16);
    dlo[idx] = (unsigned)l0 | ((unsigned)l1 << 16);
  }
}

// =====================================================================
// Kernel 2: logits[b,h,m,n] = sum_d q[m,d]*k[n,d]  (bf16x3 WMMA),
//           fused pad mask + strict-lower-triangular mask + 1/8 scale.
//   Grid: (16 tiles of 128x128, 144 bh). Wave tile 32(M) x 64(N).
//   Fragments loaded straight from global (layout matches ISA exactly).
// =====================================================================
__global__ __launch_bounds__(256) void logits_kernel(
    const unsigned* __restrict__ qhi, const unsigned* __restrict__ qlo,
    const unsigned* __restrict__ khi, const unsigned* __restrict__ klo,
    const float* __restrict__ mask,  // (B, S)
    float* __restrict__ out)         // (B, 9, S, S)
{
  const int t    = threadIdx.x;
  const int lane = t & 31;
  const int ln   = lane & 15;
  const int lh   = lane >> 4;
  const int wid  = t >> 5;
  const int wm   = wid & 3;            // 0..3 : +wm*32 in M
  const int wn   = wid >> 2;           // 0..1 : +wn*64 in N
  const int bh   = blockIdx.y;         // 0..143
  const int b    = bh / ENT;
  const int tile = blockIdx.x;         // 0..15
  const int tm   = tile >> 2;
  const int tn   = tile & 3;

  const size_t base = (size_t)bh * S_LEN * 32;   // u32 units
  const unsigned* qh = qhi + base;
  const unsigned* ql = qlo + base;
  const unsigned* kh = khi + base;
  const unsigned* kl = klo + base;

  const int m0 = tm * 128 + wm * 32;
  const int n0 = tn * 128 + wn * 64;

  v8f zero = {0.f,0.f,0.f,0.f,0.f,0.f,0.f,0.f};
  Acc acc[2][4];
#pragma unroll
  for (int i = 0; i < 2; ++i)
#pragma unroll
    for (int j = 0; j < 4; ++j) acc[i][j].v = zero;

#pragma unroll
  for (int kc = 0; kc < INNER; kc += 32) {
    Frag ah[2], al[2];
#pragma unroll
    for (int mt = 0; mt < 2; ++mt) {
      int m = m0 + mt * 16 + ln;
#pragma unroll
      for (int v = 0; v < 8; ++v) {
        int d = kc + ((v & 4) << 2) + lh * 8 + ((v & 3) << 1);
        ah[mt].u[v] = qh[m * 32 + (d >> 1)];
        al[mt].u[v] = ql[m * 32 + (d >> 1)];
      }
    }
    Frag bhf[4], blf[4];
#pragma unroll
    for (int nt = 0; nt < 4; ++nt) {
      int n = n0 + nt * 16 + ln;
#pragma unroll
      for (int v = 0; v < 8; ++v) {
        int d = kc + ((v & 4) << 2) + lh * 8 + ((v & 3) << 1);
        bhf[nt].u[v] = kh[n * 32 + (d >> 1)];
        blf[nt].u[v] = kl[n * 32 + (d >> 1)];
      }
    }
#pragma unroll
    for (int mt = 0; mt < 2; ++mt)
#pragma unroll
      for (int nt = 0; nt < 4; ++nt) {
        acc[mt][nt].v = wmma_bf16(ah[mt].v, bhf[nt].v, acc[mt][nt].v);
        acc[mt][nt].v = wmma_bf16(ah[mt].v, blf[nt].v, acc[mt][nt].v);
        acc[mt][nt].v = wmma_bf16(al[mt].v, bhf[nt].v, acc[mt][nt].v);
      }
  }

  // ---- mask + scale epilogue ----
#pragma unroll
  for (int mt = 0; mt < 2; ++mt)
#pragma unroll
    for (int nt = 0; nt < 4; ++nt) {
      int n = n0 + nt * 16 + ln;
      float pad  = mask[b * S_LEN + n];
      float pen  = -(1.0f - pad) * NEGINF;
#pragma unroll
      for (int r = 0; r < 8; ++r) {
        int m = m0 + mt * 16 + r + 8 * lh;
        float v = acc[mt][nt].f[r] * pad + pen;
        if (m > n) v -= NEGINF;                  // tril(ones, -1)
        v *= 0.125f;                             // 1/sqrt(64)
        out[((size_t)bh * S_LEN + m) * S_LEN + n] = v;
      }
    }
}

// =====================================================================
extern "C" void kernel_launch(void* const* d_in, const int* in_sizes, int n_in,
                              void* d_out, int out_size, void* d_ws, size_t ws_size,
                              hipStream_t stream)
{
  (void)in_sizes; (void)n_in; (void)out_size; (void)ws_size;
  const float* X    = (const float*)d_in[0];   // last_hidden_state (16,512,768)
  const float* mask = (const float*)d_in[1];   // attention_mask (16,512)
  const float* W    = (const float*)d_in[2];   // W_dense (768,1152)
  const float* bias = (const float*)d_in[3];   // b_dense (1152)
  float* out = (float*)d_out;                  // (16,9,512,512)

  const size_t Qu32 = (size_t)B_SZ * ENT * S_LEN * 32;  // packed bf16-pair u32s per array
  unsigned* qhi = (unsigned*)d_ws;
  unsigned* qlo = qhi + Qu32;
  unsigned* khi = qlo + Qu32;
  unsigned* klo = khi + Qu32;                  // total ws use: 4 * 9.44 MB = 37.75 MB

  proj_rope_kernel<<<dim3(18, 64), 256, 0, stream>>>(X, W, bias, qhi, qlo, khi, klo);
  logits_kernel<<<dim3(16, 144), 256, 0, stream>>>(qhi, qlo, khi, klo, mask, out);
}
